// GPT2SlidingWindowAttention_4526895530009
// MI455X (gfx1250) — matrix operand order repaired and checked by reading, not compiled, not run on hardware
//
#include <hip/hip_runtime.h>
#include <hip/hip_bf16.h>
#include <math.h>

// ---------------- types ----------------
typedef __attribute__((ext_vector_type(16))) _Float16 v16h;
typedef __attribute__((ext_vector_type(8)))  _Float16 v8h;
typedef __attribute__((ext_vector_type(8)))  float    v8f;

#define H_SIZE 768
#define N_HEADS 12
#define HEAD_DIM 64
#define SEQ 1024
#define BATCH 2
#define ROWS (BATCH*SEQ)      // 2048
#define KTILES (H_SIZE/16)    // 48

static __device__ __forceinline__ v8f wmma32(v16h a, v16h b, v8f c) {
  return __builtin_amdgcn_wmma_f32_16x16x32_f16(false, a, false, b, (short)0, c, false, false);
}

// A-fragment (16x32 f16, MxK): lane L (m=L&15, half=L>>4) holds
// K = {16*half + 0..15} contiguous in VGPR0-7.
static __device__ __forceinline__ v16h load_frag_a(const _Float16* base, int half) {
  v8h lo = *(const v8h*)(base + 16 * half);
  v8h hi = *(const v8h*)(base + 16 * half + 8);
  v16h r;
#pragma unroll
  for (int i = 0; i < 8; ++i) { r[i] = lo[i]; r[8 + i] = hi[i]; }
  return r;
}

// ---------------- phase 0: conversions ----------------
__global__ void cvt_f32_to_f16(const float* __restrict__ src, _Float16* __restrict__ dst, int n) {
  int i = blockIdx.x * 256 + threadIdx.x;
  if (i < n) dst[i] = (_Float16)src[i];
}

// Pack W (HxH row-major, f32) into f16 tiles: Wp[((nt*48+kt)*16+ni)*16+ki] = W[nt*16+ni][kt*16+ki]
// => a B-fragment lane (column ni of K-subtile kt) is 16 contiguous f16 (32B).
__global__ void pack_weight(const float* __restrict__ W, _Float16* __restrict__ Wp) {
  int i = blockIdx.x * 256 + threadIdx.x;
  if (i >= H_SIZE * H_SIZE) return;
  int ki = i & 15;
  int ni = (i >> 4) & 15;
  int t  = i >> 8;          // nt*48 + kt
  int kt = t % KTILES;
  int nt = t / KTILES;
  Wp[i] = (_Float16)W[(nt * 16 + ni) * H_SIZE + kt * 16 + ki];
}

// ---------------- GEMM: Y = X(2048x768) @ W.T + bias ----------------
// MODE 0: write f16 row-major (Q/K); MODE 1: write f16 d-major transposed per (b,h) (V);
// MODE 2: write f32 row-major (final output).
// Software-pipelined: prefetch step k+1 fragments before issuing step k WMMAs.
template <int MODE>
__global__ void __launch_bounds__(128) gemm_wmma(const _Float16* __restrict__ A,
                                                 const _Float16* __restrict__ Wp,
                                                 const float* __restrict__ bias,
                                                 void* __restrict__ out) {
  const int wid  = blockIdx.x * 4 + (threadIdx.x >> 5);
  const int lane = threadIdx.x & 31;
  const int col  = lane & 15;
  const int half = lane >> 4;
  const int mt = wid / 12;        // 0..127 (row tile)
  const int ng = wid % 12;        // 0..11  (64-col group)
  const int row0 = mt * 16;

  v8f c[4] = {};
  const _Float16* arow = A + (size_t)(row0 + col) * H_SIZE;
  const _Float16* wbase = Wp + ((size_t)(ng * 4) * KTILES * 16 + col) * 16;
  const size_t wnt_stride = (size_t)KTILES * 16 * 16;   // elems between n-tiles
  // B fragment for K-step kt2, n-tile t: wbase + t*wnt_stride + (kt2*2+half)*256

  // --- prologue: load step 0 ---
  v16h a_cur = load_frag_a(arow, half);
  v16h b_cur[4];
#pragma unroll
  for (int t = 0; t < 4; ++t)
    b_cur[t] = *(const v16h*)(wbase + t * wnt_stride + (size_t)(0 * 2 + half) * 256);

  for (int kt2 = 0; kt2 < 24; ++kt2) {
    // prefetch next step (clamped; last iteration re-loads current step harmlessly)
    const int knx = (kt2 < 23) ? kt2 + 1 : 23;
    v16h a_nxt = load_frag_a(arow + knx * 32, half);
    v16h b_nxt[4];
#pragma unroll
    for (int t = 0; t < 4; ++t)
      b_nxt[t] = *(const v16h*)(wbase + t * wnt_stride + (size_t)(knx * 2 + half) * 256);

    // compute current step on already-resident registers
#pragma unroll
    for (int t = 0; t < 4; ++t)
      c[t] = wmma32(a_cur, b_cur[t], c[t]);

    a_cur = a_nxt;
#pragma unroll
    for (int t = 0; t < 4; ++t) b_cur[t] = b_nxt[t];
  }

  if (MODE == 2) {
    float* o = (float*)out;
#pragma unroll
    for (int t = 0; t < 4; ++t) {
      const int n = ng * 64 + t * 16 + col;
      const float bv = bias[n];
#pragma unroll
      for (int r = 0; r < 8; ++r)
        o[(size_t)(row0 + r + 8 * half) * H_SIZE + n] = c[t][r] + bv;
    }
  } else if (MODE == 0) {
    _Float16* o = (_Float16*)out;
#pragma unroll
    for (int t = 0; t < 4; ++t) {
      const int n = ng * 64 + t * 16 + col;
      const float bv = bias[n];
#pragma unroll
      for (int r = 0; r < 8; ++r)
        o[(size_t)(row0 + r + 8 * half) * H_SIZE + n] = (_Float16)(c[t][r] + bv);
    }
  } else { // MODE 1: V transposed: Vt[((b*12+h)*64+d)*1024 + s]
    _Float16* o = (_Float16*)out;
    const int b  = row0 >> 10;
    const int sl = row0 & 1023;
    const int h  = ng;
#pragma unroll
    for (int t = 0; t < 4; ++t) {
      const int d = t * 16 + col;
      const float bv = bias[h * 64 + d];
      v8h pk;
#pragma unroll
      for (int r = 0; r < 8; ++r) pk[r] = (_Float16)(c[t][r] + bv);
      *(v8h*)(o + (((size_t)(b * N_HEADS + h) * 64 + d) * SEQ + sl + 8 * half)) = pk;
    }
  }
}

// ---------------- attention: per (b, h, 16-query block) ----------------
__global__ void __launch_bounds__(128) attn_wmma(const _Float16* __restrict__ Q,
                                                 const _Float16* __restrict__ K,
                                                 const _Float16* __restrict__ Vt,
                                                 _Float16* __restrict__ Ctx) {
  __shared__ __align__(16) _Float16 plds[4][16 * 96];

  const int wv   = threadIdx.x >> 5;
  const int wid  = blockIdx.x * 4 + wv;
  const int lane = threadIdx.x & 31;
  const int col  = lane & 15;
  const int half = lane >> 4;

  const int b    = wid / (N_HEADS * 64);
  const int rem  = wid % (N_HEADS * 64);
  const int h    = rem / 64;
  const int s0   = (rem % 64) * 16;
  const int jstart = s0 - 32;

  // Q fragments for the two K-steps over d=64
  const _Float16* qrow = Q + (size_t)(b * SEQ + s0 + col) * H_SIZE + h * 64;
  v16h aq0 = load_frag_a(qrow, half);
  v16h aq1 = load_frag_a(qrow + 32, half);

  // scores over 6 key tiles (96 keys: s0-32 .. s0+63)
  v8f sc[6];
#pragma unroll
  for (int jt = 0; jt < 6; ++jt) {
    int j = jstart + jt * 16 + col;
    int jcl = j < 0 ? 0 : (j > SEQ - 1 ? SEQ - 1 : j);
    const _Float16* krow = K + (size_t)(b * SEQ + jcl) * H_SIZE + h * 64;
    v16h b0 = *(const v16h*)(krow + half * 16);        // d-subtile = half
    v16h b1 = *(const v16h*)(krow + 32 + half * 16);   // d-subtile = 2 + half
    v8f z = {};
    z = wmma32(aq0, b0, z);
    z = wmma32(aq1, b1, z);
    sc[jt] = z;
  }

  // mask to reference semantics: out-of-window -> -inf; zero-padded keys -> logit 0
#pragma unroll
  for (int r = 0; r < 8; ++r) {
    const int s = s0 + r + 8 * half;
#pragma unroll
    for (int jt = 0; jt < 6; ++jt) {
      const int j = jstart + jt * 16 + col;
      float v = sc[jt][r];
      if (j < s - 32 || j > s + 31) v = -1e30f;
      else if (j < 0 || j >= SEQ)   v = 0.0f;
      else                          v *= 0.125f;   // 1/sqrt(64)
      sc[jt][r] = v;
    }
  }

  // softmax per row (row = across 16 lanes within a half) + write probs to LDS
#pragma unroll
  for (int r = 0; r < 8; ++r) {
    float mx = sc[0][r];
#pragma unroll
    for (int jt = 1; jt < 6; ++jt) mx = fmaxf(mx, sc[jt][r]);
    for (int m = 1; m <= 8; m <<= 1) mx = fmaxf(mx, __shfl_xor(mx, m, 32));
    float sum = 0.f;
#pragma unroll
    for (int jt = 0; jt < 6; ++jt) { float e = __expf(sc[jt][r] - mx); sc[jt][r] = e; sum += e; }
    for (int m = 1; m <= 8; m <<= 1) sum += __shfl_xor(sum, m, 32);
    const float inv = 1.0f / sum;
    const int srow = r + 8 * half;
#pragma unroll
    for (int jt = 0; jt < 6; ++jt) {
      const int j = jstart + jt * 16 + col;
      float p = sc[jt][r] * inv;
      if (j < 0 || j >= SEQ) p = 0.f;   // zero-padded V rows
      plds[wv][srow * 96 + jt * 16 + col] = (_Float16)p;
    }
  }
  asm volatile("s_wait_dscnt 0" ::: "memory");

  // ctx = P(16x96) x Vwin(96x64)
  v8f cc[4] = {};
  const _Float16* pbase = &plds[wv][0];
  const size_t vhead = (size_t)(b * N_HEADS + h) * 64;
#pragma unroll
  for (int ks = 0; ks < 3; ++ks) {
    v16h ap = load_frag_a(pbase + (size_t)col * 96 + ks * 32, half);
    int jb = jstart + ks * 32 + half * 16;                 // 16-aligned j group
    int jbc = jb < 0 ? 0 : (jb > SEQ - 16 ? SEQ - 16 : jb); // fully-invalid groups have P==0
#pragma unroll
    for (int dt = 0; dt < 4; ++dt) {
      const int d = dt * 16 + col;
      v16h bv = *(const v16h*)(Vt + (vhead + d) * SEQ + jbc);
      cc[dt] = wmma32(ap, bv, cc[dt]);
    }
  }

#pragma unroll
  for (int dt = 0; dt < 4; ++dt) {
    const int n = h * 64 + dt * 16 + col;
#pragma unroll
    for (int r = 0; r < 8; ++r)
      Ctx[(size_t)(b * SEQ + s0 + r + 8 * half) * H_SIZE + n] = (_Float16)cc[dt][r];
  }
}

// ---------------- launcher ----------------
extern "C" void kernel_launch(void* const* d_in, const int* in_sizes, int n_in,
                              void* d_out, int out_size, void* d_ws, size_t ws_size,
                              hipStream_t stream) {
  const float* hidden = (const float*)d_in[0];
  const float* Wq = (const float*)d_in[1];
  const float* bq = (const float*)d_in[2];
  const float* Wk = (const float*)d_in[3];
  const float* bk = (const float*)d_in[4];
  const float* Wv = (const float*)d_in[5];
  const float* bv = (const float*)d_in[6];
  const float* Wo = (const float*)d_in[7];
  const float* bo = (const float*)d_in[8];

  const size_t NX = (size_t)ROWS * H_SIZE;     // 1,572,864
  const size_t NW = (size_t)H_SIZE * H_SIZE;   //   589,824

  _Float16* Xh  = (_Float16*)d_ws;
  _Float16* Wqp = Xh  + NX;
  _Float16* Wkp = Wqp + NW;
  _Float16* Wvp = Wkp + NW;
  _Float16* Wop = Wvp + NW;
  _Float16* Qf  = Wop + NW;
  _Float16* Kf  = Qf  + NX;
  _Float16* Vt  = Kf  + NX;
  _Float16* Cx  = Vt  + NX;

  cvt_f32_to_f16<<<(NX + 255) / 256, 256, 0, stream>>>(hidden, Xh, (int)NX);
  pack_weight<<<(NW + 255) / 256, 256, 0, stream>>>(Wq, Wqp);
  pack_weight<<<(NW + 255) / 256, 256, 0, stream>>>(Wk, Wkp);
  pack_weight<<<(NW + 255) / 256, 256, 0, stream>>>(Wv, Wvp);
  pack_weight<<<(NW + 255) / 256, 256, 0, stream>>>(Wo, Wop);

  // Q, K row-major f16; V transposed (d-major)
  gemm_wmma<0><<<384, 128, 0, stream>>>(Xh, Wqp, bq, (void*)Qf);
  gemm_wmma<0><<<384, 128, 0, stream>>>(Xh, Wkp, bk, (void*)Kf);
  gemm_wmma<1><<<384, 128, 0, stream>>>(Xh, Wvp, bv, (void*)Vt);

  attn_wmma<<<384, 128, 0, stream>>>(Qf, Kf, Vt, Cx);

  gemm_wmma<2><<<384, 128, 0, stream>>>(Cx, Wop, bo, d_out);
}
